// WATTNet_89593017794632
// MI455X (gfx1250) — matrix operand order repaired and checked by reading, not compiled, not run on hardware
//
#include <hip/hip_runtime.h>
#include <math.h>

typedef __attribute__((ext_vector_type(16))) _Float16 v16h;
typedef __attribute__((ext_vector_type(8)))  _Float16 v8h;
typedef __attribute__((ext_vector_type(8)))  float    v8f;

__device__ __forceinline__ v8f wmma_f16(v16h a, v16h b, v8f c) {
  return __builtin_amdgcn_wmma_f32_16x16x32_f16(false, a, false, b, (short)0, c, false, false);
}

// Branch-free f16 fragment load. Data must be pre-staged so the k-axis is
// contiguous per row. rstride in halves; k0 multiple of 32; 16B alignment holds
// for all call sites (strides are multiples of 128 halves).
// IS_B==0 (A layout): lanes 0-15 K0-7,16-23 ; lanes 16-31 K8-15,24-31
// IS_B==1 (B layout): lanes 0-15 K0-7,16-23 ; lanes 16-31 K8-15,24-31
template <int IS_B>
__device__ __forceinline__ v16h load_frag_h(const _Float16* base, long rstride, int k0) {
  const int lane = threadIdx.x & 31;
  const int r = lane & 15, half = lane >> 4;
  const _Float16* rp = base + (long)r * rstride + k0;
  v8h lo, hi;
  if (IS_B) {
    lo = *(const v8h*)(rp + 8 * half);
    hi = *(const v8h*)(rp + 8 * half + 16);
  } else {
    lo = *(const v8h*)(rp + 8 * half);
    hi = *(const v8h*)(rp + 8 * half + 16);
  }
  return __builtin_shufflevector(lo, hi, 0, 1, 2, 3, 4, 5, 6, 7, 8, 9, 10, 11, 12, 13, 14, 15);
}

// ============ weight staging kernels (run once per layer, tiny) ============

// plain f32 -> f16 copy
__global__ void k_cvt(const float* __restrict__ s, _Float16* __restrict__ d, int n) {
  int i = blockIdx.x * blockDim.x + threadIdx.x;
  if (i < n) d[i] = (_Float16)s[i];
}

// wf (128,128,2) -> wfh[o][tap*128+i]
__global__ void k_wfg(const float* __restrict__ w, _Float16* __restrict__ wh) {
  int i = blockIdx.x * blockDim.x + threadIdx.x;
  if (i >= 128 * 256) return;
  int o = i >> 8, rr = i & 255, tap = rr >> 7, ch = rr & 127;
  wh[i] = (_Float16)w[(o * 128 + ch) * 2 + tap];
}

// ws (Tin,Tout) -> wsT[s][t] zero-padded to 128x128
__global__ void k_tws(const float* __restrict__ ws, _Float16* __restrict__ wst,
                      int Tin, int Tout) {
  int i = blockIdx.x * blockDim.x + threadIdx.x;
  if (i >= 128 * 128) return;
  int s = i >> 7, t = i & 127;
  wst[i] = (s < Tout && t < Tin) ? (_Float16)ws[t * Tout + s] : (_Float16)0.f;
}

// wv (128,2048) -> wvT[l][m]
__global__ void k_twv(const float* __restrict__ wv, _Float16* __restrict__ wvt) {
  int i = blockIdx.x * blockDim.x + threadIdx.x;
  if (i >= 2048 * 128) return;
  int l = i >> 7, m = i & 127;
  wvt[i] = (_Float16)wv[m * 2048 + l];
}

// ============ embed: write XT[b][n][e][t] and XC[b][t][e][n] (f16) ============
__global__ void k_embed(const float* __restrict__ xin, const float* __restrict__ ew,
                        const float* __restrict__ eb, _Float16* __restrict__ xt,
                        _Float16* __restrict__ xc) {
  long idx = (long)blockIdx.x * blockDim.x + threadIdx.x;
  if (idx >= (long)4 * 128 * 128 * 16) return;
  int e = idx & 15;
  long r = idx >> 4;
  int t = r % 128; r /= 128;
  int n = r % 128;
  int b = (int)(r / 128);
  const float* row = xin + ((long)b * 128 + t) * 131;
  float v = row[n] * ew[e] + row[128] * ew[16 + e] + row[129] * ew[32 + e] +
            row[130] * ew[48 + e] + eb[e];
  xt[((long)(b * 128 + n) * 16 + e) * 128 + t] = (_Float16)v;
  xc[((long)(b * 128 + t) * 16 + e) * 128 + n] = (_Float16)v;  // Tin = 128
}

// ============ res[b][c][s][e] = sum_t XT[b][c][e][t] * wsT[s][t] + bs[s] ============
__global__ void k_res(const _Float16* __restrict__ xt, const _Float16* __restrict__ wst,
                      const float* __restrict__ bs, float* __restrict__ res, int Tout) {
  int tid = blockIdx.x;
  int st = tid & 7; tid >>= 3;
  int c = tid & 127;
  int b = tid >> 7;
  const _Float16* A = xt + ((long)(b * 128 + c) * 16) * 128;  // (e,t), rstride 128
  const _Float16* B = wst + (long)st * 16 * 128;              // (s_local,t), rstride 128
  v8f acc = {};
#pragma unroll
  for (int k0 = 0; k0 < 128; k0 += 32) {
    v16h a = load_frag_h<0>(A, 128, k0);
    v16h bf = load_frag_h<1>(B, 128, k0);
    acc = wmma_f16(a, bf, acc);
  }
  const int lane = threadIdx.x & 31;
  const int n = lane & 15, half = lane >> 4;
  int s = st * 16 + n;
  if (s < Tout) {
    float bsv = bs[s];
    float* out = res + (((long)(b * 128 + c)) * Tout + s) * 16;
#pragma unroll
    for (int r = 0; r < 8; ++r) out[r + 8 * half] = acc[r] + bsv;
  }
}

// ============ gated: K=256 (k = i + 128*tap); writes f32 + both mirrors ============
__global__ void k_gated(const _Float16* __restrict__ xc, const _Float16* __restrict__ wfh,
                        const _Float16* __restrict__ wgh, const float* __restrict__ bfv,
                        const float* __restrict__ bgv, const float* __restrict__ res,
                        float* __restrict__ xf, _Float16* __restrict__ xto,
                        _Float16* __restrict__ xco, int Tin, int d) {
  int Tout = Tin - d;
  int tid = blockIdx.x;
  int s = tid % Tout; tid /= Tout;
  int ot = tid & 7;
  int b = tid >> 3;
  int o0 = ot * 16;
  const _Float16* wfb = wfh + (long)o0 * 256;
  const _Float16* wgb = wgh + (long)o0 * 256;
  v8f cf = {}, cg = {};
#pragma unroll
  for (int k0 = 0; k0 < 256; k0 += 32) {
    int tap = k0 >> 7;
    const _Float16* xb = xc + ((long)(b * Tin + s + tap * d) * 16) * 128;  // (e,i)
    v16h bb = load_frag_h<1>(xb, 128, k0 & 127);
    v16h af = load_frag_h<0>(wfb, 256, k0);
    v16h ag = load_frag_h<0>(wgb, 256, k0);
    cf = wmma_f16(af, bb, cf);
    cg = wmma_f16(ag, bb, cg);
  }
  const int lane = threadIdx.x & 31;
  const int e = lane & 15, half = lane >> 4;
#pragma unroll
  for (int rr = 0; rr < 8; ++rr) {
    int o = o0 + rr + 8 * half;
    float fv = tanhf(cf[rr] + bfv[o]);
    float gv = cg[rr] + bgv[o];
    gv = 1.f / (1.f + expf(-gv));
    long oi = (((long)(b * 128 + o)) * Tout + s) * 16 + e;
    float val = fv * gv + res[oi];
    xf[oi] = val;
    xto[((long)(b * 128 + o) * 16 + e) * 128 + s] = (_Float16)val;
    xco[((long)(b * Tout + s) * 16 + e) * 128 + o] = (_Float16)val;
  }
}

// ============ Mt[m][n] = sum_l wk[m][l]*wq[n][l]  (f16 out, 64 waves) ============
__global__ void k_prepm(const _Float16* __restrict__ wkh, const _Float16* __restrict__ wqh,
                        _Float16* __restrict__ mt) {
  int tid = blockIdx.x;
  int ct = tid & 7, rt = tid >> 3;
  const _Float16* A = wkh + (long)rt * 16 * 2048;
  const _Float16* B = wqh + (long)ct * 16 * 2048;
  v8f acc = {};
#pragma unroll 4
  for (int k0 = 0; k0 < 2048; k0 += 32) {
    v16h a = load_frag_h<0>(A, 2048, k0);
    v16h b = load_frag_h<1>(B, 2048, k0);
    acc = wmma_f16(a, b, acc);
  }
  const int lane = threadIdx.x & 31;
  const int n = lane & 15, half = lane >> 4;
#pragma unroll
  for (int r = 0; r < 8; ++r)
    mt[(long)(rt * 16 + r + 8 * half) * 128 + ct * 16 + n] = (_Float16)acc[r];
}

// ============ uq = wq@bk, uk = wk@bq, cqk = bq·bk (f32) ============
__global__ void k_prepvec(const float* __restrict__ wq, const float* __restrict__ wk,
                          const float* __restrict__ bq, const float* __restrict__ bk,
                          float* __restrict__ uq, float* __restrict__ uk,
                          float* __restrict__ cqk) {
  int t = threadIdx.x;
  if (t < 128) {
    float s = 0.f; const float* row = wq + (long)t * 2048;
    for (int l = 0; l < 2048; ++l) s += row[l] * bk[l];
    uq[t] = s;
  } else {
    int m = t - 128;
    float s = 0.f; const float* row = wk + (long)m * 2048;
    for (int l = 0; l < 2048; ++l) s += row[l] * bq[l];
    uk[m] = s;
    if (m == 0) {
      float c = 0.f;
      for (int l = 0; l < 2048; ++l) c += bq[l] * bk[l];
      *cqk = c;
    }
  }
}

// ============ per-(b,t): S = scale*(X·Mqk·X^T + biases), softmax, abar, z ============
__global__ void k_score(const _Float16* __restrict__ xc, const _Float16* __restrict__ mt,
                        const float* __restrict__ uq, const float* __restrict__ uk,
                        const float* __restrict__ cqkp, _Float16* __restrict__ zbh, int T) {
  __shared__ __align__(16) _Float16 Pl[16 * 128];
  __shared__ float Sl[16 * 16];
  __shared__ float al[16], bl[16], abar[16];
  int bt = blockIdx.x;
  int b = bt / T, t = bt % T;
  const _Float16* xb = xc + ((long)(b * T + t) * 16) * 128;  // X[e][n] = xb[e*128+n]
  const int lane = threadIdx.x & 31;
  const int nl = lane & 15, half = lane >> 4;

  v8f zero = {};
  v8f pacc[8];
#pragma unroll
  for (int i = 0; i < 8; ++i) pacc[i] = zero;
  // P = X @ Mqk : B element(col m, k n) = Mqk[n][m] = mt[m*128+n]
#pragma unroll
  for (int k0 = 0; k0 < 128; k0 += 32) {
    v16h a = load_frag_h<0>(xb, 128, k0);
#pragma unroll
    for (int m8 = 0; m8 < 8; ++m8) {
      v16h bf = load_frag_h<1>(mt + (long)m8 * 16 * 128, 128, k0);
      pacc[m8] = wmma_f16(a, bf, pacc[m8]);
    }
  }
#pragma unroll
  for (int m8 = 0; m8 < 8; ++m8)
#pragma unroll
    for (int rr = 0; rr < 8; ++rr)
      Pl[(rr + 8 * half) * 128 + m8 * 16 + nl] = (_Float16)pacc[m8][rr];
  // bias dot products (contiguous f16 rows of X)
  {
    const _Float16* xr = xb + (long)nl * 128;
    const float* u = (half == 0) ? uq : uk;
    float s = 0.f;
    for (int n = 0; n < 128; ++n) s += (float)xr[n] * u[n];
    if (half == 0) al[nl] = s; else bl[nl] = s;
  }
  __syncthreads();
  // S = P @ X^T
  v8f sacc = zero;
#pragma unroll
  for (int k0 = 0; k0 < 128; k0 += 32) {
    v16h a = load_frag_h<0>(Pl, 128, k0);
    v16h bf = load_frag_h<1>(xb, 128, k0);
    sacc = wmma_f16(a, bf, sacc);
  }
  float cq = *cqkp;
  const float scale = 0.022097086912079608f;  // 1/sqrt(2048)
#pragma unroll
  for (int rr = 0; rr < 8; ++rr) {
    int e = rr + 8 * half;
    Sl[e * 16 + nl] = (sacc[rr] + al[e] + bl[nl] + cq) * scale;
  }
  __syncthreads();
  if (half == 0) {  // row softmax
    int e = nl;
    float mx = -1e30f;
    for (int f = 0; f < 16; ++f) mx = fmaxf(mx, Sl[e * 16 + f]);
    float tmp[16]; float sm = 0.f;
    for (int f = 0; f < 16; ++f) { tmp[f] = expf(Sl[e * 16 + f] - mx); sm += tmp[f]; }
    float inv = 1.f / sm;
    for (int f = 0; f < 16; ++f) Sl[e * 16 + f] = tmp[f] * inv;
  }
  __syncthreads();
  if (half == 0) {  // abar[f] = mean_e alpha[e][f]
    int f = nl;
    float s = 0.f;
    for (int e = 0; e < 16; ++e) s += Sl[e * 16 + f];
    abar[f] = s * 0.0625f;
  }
  __syncthreads();
  float ab[16];
#pragma unroll
  for (int f = 0; f < 16; ++f) ab[f] = abar[f];
#pragma unroll
  for (int j = 0; j < 4; ++j) {  // z[m] = sum_f abar[f]*X[f][m]
    int m = lane * 4 + j;
    float s = 0.f;
#pragma unroll
    for (int f = 0; f < 16; ++f) s += ab[f] * (float)xb[(long)f * 128 + m];
    zbh[(long)bt * 128 + m] = (_Float16)s;
  }
}

// ============ read: x += zb @ wv + bv (fused residual, refresh mirrors) ============
__global__ void k_read(const _Float16* __restrict__ zbh, const _Float16* __restrict__ wvt,
                       const float* __restrict__ bv, float* __restrict__ xf,
                       _Float16* __restrict__ xt, _Float16* __restrict__ xc, int T) {
  int BT = 4 * T;
  int tid = blockIdx.x;
  int nt = tid & 127;
  int mt = tid >> 7;
  int m0 = mt * 16;
  const _Float16* A = zbh + (long)m0 * 128;          // (bt, k=ch)
  const _Float16* B = wvt + (long)nt * 16 * 128;     // (l_local, k=ch)
  v8f acc = {};
#pragma unroll
  for (int k0 = 0; k0 < 128; k0 += 32) {
    v16h a = load_frag_h<0>(A, 128, k0);
    v16h b = load_frag_h<1>(B, 128, k0);
    acc = wmma_f16(a, b, acc);
  }
  const int lane = threadIdx.x & 31;
  const int e = lane & 15, half = lane >> 4;
  float bvv = bv[nt * 16 + e];
#pragma unroll
  for (int rr = 0; rr < 8; ++rr) {
    int bt = m0 + rr + 8 * half;
    if (bt < BT) {
      int b = bt / T, t = bt % T;
      long oi = (((long)(b * 128 + nt)) * T + t) * 16 + e;
      float val = xf[oi] + acc[rr] + bvv;
      xf[oi] = val;
      xt[((long)(b * 128 + nt) * 16 + e) * 128 + t] = (_Float16)val;
      xc[((long)(b * T + t) * 16 + e) * 128 + nt] = (_Float16)val;
    }
  }
}

// ============ decode ============
__global__ void k_dec(const float* __restrict__ x2, const float* __restrict__ dw,
                      const float* __restrict__ db, float* __restrict__ y, int T) {
  long idx = (long)blockIdx.x * blockDim.x + threadIdx.x;
  long total = (long)4 * 128 * T;
  if (idx >= total) return;
  int t = (int)(idx % T);
  long r = idx / T;
  int n = r % 128;
  int b = (int)(r / 128);
  const float* xr = x2 + (((long)(b * 128 + n)) * T + t) * 16;
  const float* wr = dw + n * 16;
  float s = db[n];
#pragma unroll
  for (int e = 0; e < 16; ++e) s += xr[e] * wr[e];
  y[idx] = s;
}

__global__ void k_out(const float* __restrict__ y, const float* __restrict__ ow,
                      const float* __restrict__ ob, float* __restrict__ out, int T) {
  int idx = blockIdx.x * blockDim.x + threadIdx.x;
  if (idx >= 4 * 24 * 128) return;
  int n = idx & 127;
  int r = idx >> 7;
  int o = r % 24;
  int b = r / 24;
  const float* yr = y + ((long)(b * 128 + n)) * T;
  float s = ob[o];
  for (int t = 0; t < T; ++t) s += yr[t] * ow[t * 24 + o];
  out[idx] = s;
}

extern "C" void kernel_launch(void* const* d_in, const int* in_sizes, int n_in,
                              void* d_out, int out_size, void* d_ws, size_t ws_size,
                              hipStream_t stream) {
  (void)in_sizes; (void)n_in; (void)out_size; (void)ws_size;
  const float* x_in  = (const float*)d_in[0];
  const float* emb_w = (const float*)d_in[1];
  const float* emb_b = (const float*)d_in[2];
  const float* gwf[2] = {(const float*)d_in[3],  (const float*)d_in[15]};
  const float* gbf[2] = {(const float*)d_in[4],  (const float*)d_in[16]};
  const float* gwg[2] = {(const float*)d_in[5],  (const float*)d_in[17]};
  const float* gbg[2] = {(const float*)d_in[6],  (const float*)d_in[18]};
  const float* gws[2] = {(const float*)d_in[7],  (const float*)d_in[19]};
  const float* gbs[2] = {(const float*)d_in[8],  (const float*)d_in[20]};
  const float* awq[2] = {(const float*)d_in[9],  (const float*)d_in[21]};
  const float* abq[2] = {(const float*)d_in[10], (const float*)d_in[22]};
  const float* awk[2] = {(const float*)d_in[11], (const float*)d_in[23]};
  const float* abk[2] = {(const float*)d_in[12], (const float*)d_in[24]};
  const float* awv[2] = {(const float*)d_in[13], (const float*)d_in[25]};
  const float* abv[2] = {(const float*)d_in[14], (const float*)d_in[26]};
  const float* dec_w = (const float*)d_in[27];
  const float* dec_b = (const float*)d_in[28];
  const float* out_w = (const float*)d_in[29];
  const float* out_b = (const float*)d_in[30];

  // ---- workspace carve-up
  float* ws = (float*)d_ws;
  float* X1f = ws;                  // 4*128*126*16 = 1032192
  float* X2f = X1f + 1032192;       // 999424
  float* RES = X2f + 999424;        // 1032192
  float* UQ  = RES + 1032192;       // 128
  float* UK  = UQ + 128;            // 128
  float* CQK = UK + 128;            // 128 (pad)
  float* Y   = CQK + 128;           // 62464
  _Float16* h = (_Float16*)(Y + 62464);
  _Float16* XT  = h;                // 4*128*16*128 = 1048576
  _Float16* XCa = XT + 1048576;     // 1048576
  _Float16* XCb = XCa + 1048576;    // 1048576
  _Float16* WQH = XCb + 1048576;    // 262144
  _Float16* WKH = WQH + 262144;     // 262144
  _Float16* WVT = WKH + 262144;     // 262144
  _Float16* WFH = WVT + 262144;     // 32768
  _Float16* WGH = WFH + 32768;      // 32768
  _Float16* WST = WGH + 32768;      // 16384
  _Float16* MT  = WST + 16384;      // 16384
  _Float16* ZBH = MT + 16384;       // 65536

  const int Tin_[2]  = {128, 126};
  const int Tout_[2] = {126, 122};
  const int dil_[2]  = {2, 4};
  float* Xf_[2] = {X1f, X2f};

  // ---- embed -> XT + XCa (T=128)
  const long NE = (long)4 * 128 * 128 * 16;
  k_embed<<<dim3((unsigned)((NE + 255) / 256)), dim3(256), 0, stream>>>(x_in, emb_w, emb_b, XT, XCa);

  _Float16* XCin = XCa;
  _Float16* XCout = XCb;
  for (int l = 0; l < 2; ++l) {
    int Tin = Tin_[l], Tout = Tout_[l], d = dil_[l];
    float* Xf = Xf_[l];
    // stage weights
    k_tws<<<dim3(64), dim3(256), 0, stream>>>(gws[l], WST, Tin, Tout);
    k_wfg<<<dim3(128), dim3(256), 0, stream>>>(gwf[l], WFH);
    k_wfg<<<dim3(128), dim3(256), 0, stream>>>(gwg[l], WGH);
    k_cvt<<<dim3(1024), dim3(256), 0, stream>>>(awq[l], WQH, 262144);
    k_cvt<<<dim3(1024), dim3(256), 0, stream>>>(awk[l], WKH, 262144);
    k_twv<<<dim3(1024), dim3(256), 0, stream>>>(awv[l], WVT);
    // res (reads XT), then clear XT pad, then gated writes XT/XCout/Xf
    k_res<<<dim3(4 * 128 * 8), dim3(32), 0, stream>>>(XT, WST, gbs[l], RES, Tout);
    hipMemsetAsync(XT, 0, (size_t)1048576 * sizeof(_Float16), stream);
    k_gated<<<dim3(4 * 8 * Tout), dim3(32), 0, stream>>>(XCin, WFH, WGH, gbf[l], gbg[l],
                                                         RES, Xf, XT, XCout, Tin, d);
    // attention
    k_prepm<<<dim3(64), dim3(32), 0, stream>>>(WKH, WQH, MT);
    k_prepvec<<<dim3(1), dim3(256), 0, stream>>>(awq[l], awk[l], abq[l], abk[l], UQ, UK, CQK);
    k_score<<<dim3(4 * Tout), dim3(32), 0, stream>>>(XCout, MT, UQ, UK, CQK, ZBH, Tout);
    k_read<<<dim3(((4 * Tout + 15) / 16) * 128), dim3(32), 0, stream>>>(ZBH, WVT, abv[l],
                                                                        Xf, XT, XCout, Tout);
    _Float16* tmp = XCin; XCin = XCout; XCout = tmp;
  }

  // ---- decode
  const long NY = (long)4 * 128 * 122;
  k_dec<<<dim3((unsigned)((NY + 255) / 256)), dim3(256), 0, stream>>>(X2f, dec_w, dec_b, Y, 122);
  k_out<<<dim3((4 * 24 * 128 + 255) / 256), dim3(256), 0, stream>>>(Y, out_w, out_b,
                                                                    (float*)d_out, 122);
}
